// LingoAuraLLM_91207925498292
// MI455X (gfx1250) — matrix operand order repaired and checked by reading, not compiled, not run on hardware
//
#include <hip/hip_runtime.h>
#include <hip/hip_bf16.h>
#include <hip/hip_fp16.h>

typedef __attribute__((ext_vector_type(16))) _Float16 v16h;
typedef __attribute__((ext_vector_type(8)))  _Float16 h8;
typedef __attribute__((ext_vector_type(2)))  _Float16 h2;
typedef __attribute__((ext_vector_type(8)))  float    v8f;

#define ACT_NONE 0
#define ACT_RELU 1

#define WMMA_F16(a, b, c) \
    __builtin_amdgcn_wmma_f32_16x16x32_f16(false, (a), false, (b), (short)0, (c), false, false)

// ---------------------------------------------------------------------------
// WMMA GEMM: C[M,N] (+)= act(alpha * A[M,K]@B[K,N] + bias)
// 256 threads = 8 waves; block tile 64(M) x 128(N), K-step 32.
// Wave grid 2x4; each wave computes 32x32 = 2 A-frags x 2 B-frags = 4 WMMAs.
// LDS: As[64][40] row-major f16, BsT[128][40] transposed B (b128 frag reads).
// Interior blocks: guard-free loop, next K-tile register-staged under compute.
// Edge blocks: branch-free clamped loads.
// ---------------------------------------------------------------------------
__global__ __launch_bounds__(256)
void gemm_wmma_kernel(const float* __restrict__ A, int lda,
                      const float* __restrict__ Bm, int ldb,
                      const float* __restrict__ bias,
                      float* __restrict__ C, int ldc,
                      int M, int N, int K,
                      float alpha, int act, int accum)
{
    __shared__ alignas(16) _Float16 As[64][40];
    __shared__ alignas(16) _Float16 BsT[128][40];
    const int bm = blockIdx.y * 64;
    const int bn = blockIdx.x * 128;
    const int tid = threadIdx.x;
    const int lane = tid & 31;
    const int wave = tid >> 5;
    const int wm  = (wave >> 2) * 32;    // 0 / 32
    const int wnn = (wave & 3) * 32;     // 0 / 32 / 64 / 96
    const int mlane = lane & 15;
    const int koff = (lane < 16) ? 0 : 8;   // A frag K sub-offset
    const int kbb  = (lane < 16) ? 0 : 16;  // unused; B frags are read at koff like A

    const int ar = tid >> 2;            // A fill: row 0..63
    const int ac = (tid & 3) * 8;       // A fill: col 0,8,16,24
    const int k2 = (tid >> 4) * 2;      // B fill: k pair 0..30
    const int c8 = (tid & 15) * 8;      // B fill: col group 0..120

    v8f c00 = {}, c01 = {}, c10 = {}, c11 = {};

    auto compute_step = [&]() {
        h8 alo0 = *(const h8*)&As[wm + mlane][koff];
        h8 ahi0 = *(const h8*)&As[wm + mlane][16 + koff];
        h8 alo1 = *(const h8*)&As[wm + 16 + mlane][koff];
        h8 ahi1 = *(const h8*)&As[wm + 16 + mlane][16 + koff];
        h8 b0lo = *(const h8*)&BsT[wnn + mlane][koff];
        h8 b0hi = *(const h8*)&BsT[wnn + mlane][16 + koff];
        h8 b1lo = *(const h8*)&BsT[wnn + 16 + mlane][koff];
        h8 b1hi = *(const h8*)&BsT[wnn + 16 + mlane][16 + koff];
        v16h a0 = __builtin_shufflevector(alo0, ahi0, 0,1,2,3,4,5,6,7,8,9,10,11,12,13,14,15);
        v16h a1 = __builtin_shufflevector(alo1, ahi1, 0,1,2,3,4,5,6,7,8,9,10,11,12,13,14,15);
        v16h b0 = __builtin_shufflevector(b0lo, b0hi, 0,1,2,3,4,5,6,7,8,9,10,11,12,13,14,15);
        v16h b1 = __builtin_shufflevector(b1lo, b1hi, 0,1,2,3,4,5,6,7,8,9,10,11,12,13,14,15);
        c00 = WMMA_F16(a0, b0, c00);
        c01 = WMMA_F16(a0, b1, c01);
        c10 = WMMA_F16(a1, b0, c10);
        c11 = WMMA_F16(a1, b1, c11);
    };

    const bool fastblock = (bm + 64 <= M) && (bn + 128 <= N) && ((K & 31) == 0);

    if (fastblock) {
        float4 aF0, aF1, bR0a, bR0b, bR1a, bR1b;
        auto load_regs = [&](int kk) {
            const float* srcA = A + (long)(bm + ar) * lda + kk + ac;
            aF0 = *(const float4*)(srcA);
            aF1 = *(const float4*)(srcA + 4);
            const float* s0 = Bm + (long)(kk + k2) * ldb + bn + c8;
            bR0a = *(const float4*)(s0);
            bR0b = *(const float4*)(s0 + 4);
            bR1a = *(const float4*)(s0 + ldb);
            bR1b = *(const float4*)(s0 + ldb + 4);
        };
        auto store_lds = [&]() {
            h8 hv;
            hv[0] = (_Float16)aF0.x; hv[1] = (_Float16)aF0.y;
            hv[2] = (_Float16)aF0.z; hv[3] = (_Float16)aF0.w;
            hv[4] = (_Float16)aF1.x; hv[5] = (_Float16)aF1.y;
            hv[6] = (_Float16)aF1.z; hv[7] = (_Float16)aF1.w;
            *(h8*)&As[ar][ac] = hv;
            h2 p;
            p[0] = (_Float16)bR0a.x; p[1] = (_Float16)bR1a.x; *(h2*)&BsT[c8 + 0][k2] = p;
            p[0] = (_Float16)bR0a.y; p[1] = (_Float16)bR1a.y; *(h2*)&BsT[c8 + 1][k2] = p;
            p[0] = (_Float16)bR0a.z; p[1] = (_Float16)bR1a.z; *(h2*)&BsT[c8 + 2][k2] = p;
            p[0] = (_Float16)bR0a.w; p[1] = (_Float16)bR1a.w; *(h2*)&BsT[c8 + 3][k2] = p;
            p[0] = (_Float16)bR0b.x; p[1] = (_Float16)bR1b.x; *(h2*)&BsT[c8 + 4][k2] = p;
            p[0] = (_Float16)bR0b.y; p[1] = (_Float16)bR1b.y; *(h2*)&BsT[c8 + 5][k2] = p;
            p[0] = (_Float16)bR0b.z; p[1] = (_Float16)bR1b.z; *(h2*)&BsT[c8 + 6][k2] = p;
            p[0] = (_Float16)bR0b.w; p[1] = (_Float16)bR1b.w; *(h2*)&BsT[c8 + 7][k2] = p;
        };
        load_regs(0);
        for (int k0 = 0; k0 < K; k0 += 32) {
            store_lds();
            __syncthreads();
            if (k0 + 32 < K) {
                load_regs(k0 + 32);   // global loads overlap with WMMAs below
                if (k0 + 64 < K)
                    __builtin_prefetch(&Bm[(long)(k0 + 64 + k2) * ldb + bn + c8], 0, 1);
            }
            compute_step();
            __syncthreads();
        }
    } else {
        for (int k0 = 0; k0 < K; k0 += 32) {
            // branch-free clamped fills
            int gr = bm + ar;
            int grc = gr < M - 1 ? gr : M - 1;
            h8 hv;
            #pragma unroll
            for (int e = 0; e < 8; ++e) {
                int gc = k0 + ac + e;
                int gcc = gc < K - 1 ? gc : K - 1;
                float v = A[(long)grc * lda + gcc];
                hv[e] = (_Float16)((gr < M && gc < K) ? v : 0.0f);
            }
            *(h8*)&As[ar][ac] = hv;

            int kr0 = k0 + k2, kr1 = k0 + k2 + 1;
            int kc0 = kr0 < K - 1 ? kr0 : K - 1;
            int kc1 = kr1 < K - 1 ? kr1 : K - 1;
            #pragma unroll
            for (int i = 0; i < 8; ++i) {
                int gc = bn + c8 + i;
                int gcc = gc < N - 1 ? gc : N - 1;
                float v0 = Bm[(long)kc0 * ldb + gcc];
                float v1 = Bm[(long)kc1 * ldb + gcc];
                h2 p;
                p[0] = (_Float16)((kr0 < K && gc < N) ? v0 : 0.0f);
                p[1] = (_Float16)((kr1 < K && gc < N) ? v1 : 0.0f);
                *(h2*)&BsT[c8 + i][k2] = p;
            }
            __syncthreads();
            compute_step();
            __syncthreads();
        }
    }

    // epilogue
    const int rowadd = (lane >= 16) ? 8 : 0;
    auto emit = [&](int gm, int gn, float v) {
        if (gm < M && gn < N) {
            v *= alpha;
            if (bias) v += bias[gn];
            if (act == ACT_RELU) v = v > 0.0f ? v : 0.0f;
            long o = (long)gm * ldc + gn;
            if (accum) C[o] += v; else C[o] = v;
        }
    };
    #pragma unroll
    for (int r = 0; r < 8; ++r) {
        int gm0 = bm + wm + r + rowadd;
        int gm1 = bm + wm + 16 + r + rowadd;
        int gn0 = bn + wnn + mlane;
        int gn1 = bn + wnn + 16 + mlane;
        emit(gm0, gn0, c00[r]);
        emit(gm0, gn1, c01[r]);
        emit(gm1, gn0, c10[r]);
        emit(gm1, gn1, c11[r]);
    }
}

// ---------------------------------------------------------------------------
// Fused LM head + cross entropy, streaming over V=32000 (250 chunks of 128).
// 2048 valid rows (b in 0..3, t in 1..512). Never materializes logits.
// ---------------------------------------------------------------------------
__global__ __launch_bounds__(256)
void lmhead_loss_kernel(const float* __restrict__ xf,      // [2056,1024] rmsnormed
                        const float* __restrict__ W,       // [1024,32000]
                        const int*   __restrict__ labels,  // [4,512] flat
                        float* __restrict__ loss_accum)
{
    __shared__ alignas(16) _Float16 As[64][40];
    __shared__ alignas(16) _Float16 BsT[128][40];
    __shared__ float tile[64][132];
    __shared__ float rmax[64], rsum[64], rlab[64];

    const int r0 = blockIdx.x * 64;      // grid = 32 -> 2048 valid rows
    const int tid = threadIdx.x;
    const int lane = tid & 31;
    const int wave = tid >> 5;
    const int wm  = (wave >> 2) * 32;
    const int wnn = (wave & 3) * 32;
    const int mlane = lane & 15;
    const int koff = (lane < 16) ? 0 : 8;
    const int kbb  = (lane < 16) ? 0 : 16;
    const int rowadd = (lane >= 16) ? 8 : 0;

    const int ar = tid >> 2;
    const int ac = (tid & 3) * 8;
    const int k2 = (tid >> 4) * 2;
    const int c8 = (tid & 15) * 8;

    // valid-row remap for this thread's A row: g -> row b*514 + 1 + (g&511)
    const int g = r0 + ar;
    const long arow = (long)((g >> 9) * 514 + 1 + (g & 511));

    if (tid < 64) { rmax[tid] = -1e30f; rsum[tid] = 0.0f; rlab[tid] = 0.0f; }
    __syncthreads();

    for (int nc = 0; nc < 250; ++nc) {
        const int bn = nc * 128;
        v8f c00 = {}, c01 = {}, c10 = {}, c11 = {};
        for (int k0 = 0; k0 < 1024; k0 += 32) {
            const float* srcA = xf + arow * 1024 + k0 + ac;
            float4 f0 = *(const float4*)(srcA);
            float4 f1 = *(const float4*)(srcA + 4);
            h8 hv;
            hv[0] = (_Float16)f0.x; hv[1] = (_Float16)f0.y;
            hv[2] = (_Float16)f0.z; hv[3] = (_Float16)f0.w;
            hv[4] = (_Float16)f1.x; hv[5] = (_Float16)f1.y;
            hv[6] = (_Float16)f1.z; hv[7] = (_Float16)f1.w;
            *(h8*)&As[ar][ac] = hv;

            const float* s0 = W + (long)(k0 + k2) * 32000 + bn + c8;
            float4 r0a = *(const float4*)(s0);
            float4 r0b = *(const float4*)(s0 + 4);
            float4 r1a = *(const float4*)(s0 + 32000);
            float4 r1b = *(const float4*)(s0 + 32000 + 4);
            h2 p;
            p[0] = (_Float16)r0a.x; p[1] = (_Float16)r1a.x; *(h2*)&BsT[c8 + 0][k2] = p;
            p[0] = (_Float16)r0a.y; p[1] = (_Float16)r1a.y; *(h2*)&BsT[c8 + 1][k2] = p;
            p[0] = (_Float16)r0a.z; p[1] = (_Float16)r1a.z; *(h2*)&BsT[c8 + 2][k2] = p;
            p[0] = (_Float16)r0a.w; p[1] = (_Float16)r1a.w; *(h2*)&BsT[c8 + 3][k2] = p;
            p[0] = (_Float16)r0b.x; p[1] = (_Float16)r1b.x; *(h2*)&BsT[c8 + 4][k2] = p;
            p[0] = (_Float16)r0b.y; p[1] = (_Float16)r1b.y; *(h2*)&BsT[c8 + 5][k2] = p;
            p[0] = (_Float16)r0b.z; p[1] = (_Float16)r1b.z; *(h2*)&BsT[c8 + 6][k2] = p;
            p[0] = (_Float16)r0b.w; p[1] = (_Float16)r1b.w; *(h2*)&BsT[c8 + 7][k2] = p;
            __syncthreads();

            h8 alo0 = *(const h8*)&As[wm + mlane][koff];
            h8 ahi0 = *(const h8*)&As[wm + mlane][16 + koff];
            h8 alo1 = *(const h8*)&As[wm + 16 + mlane][koff];
            h8 ahi1 = *(const h8*)&As[wm + 16 + mlane][16 + koff];
            h8 b0lo = *(const h8*)&BsT[wnn + mlane][koff];
            h8 b0hi = *(const h8*)&BsT[wnn + mlane][16 + koff];
            h8 b1lo = *(const h8*)&BsT[wnn + 16 + mlane][koff];
            h8 b1hi = *(const h8*)&BsT[wnn + 16 + mlane][16 + koff];
            v16h a0 = __builtin_shufflevector(alo0, ahi0, 0,1,2,3,4,5,6,7,8,9,10,11,12,13,14,15);
            v16h a1 = __builtin_shufflevector(alo1, ahi1, 0,1,2,3,4,5,6,7,8,9,10,11,12,13,14,15);
            v16h b0 = __builtin_shufflevector(b0lo, b0hi, 0,1,2,3,4,5,6,7,8,9,10,11,12,13,14,15);
            v16h b1 = __builtin_shufflevector(b1lo, b1hi, 0,1,2,3,4,5,6,7,8,9,10,11,12,13,14,15);
            c00 = WMMA_F16(a0, b0, c00);
            c01 = WMMA_F16(a0, b1, c01);
            c10 = WMMA_F16(a1, b0, c10);
            c11 = WMMA_F16(a1, b1, c11);
            __syncthreads();
        }
        #pragma unroll
        for (int r = 0; r < 8; ++r) {
            tile[wm + r + rowadd][wnn + mlane]           = c00[r];
            tile[wm + r + rowadd][wnn + 16 + mlane]      = c01[r];
            tile[wm + 16 + r + rowadd][wnn + mlane]      = c10[r];
            tile[wm + 16 + r + rowadd][wnn + 16 + mlane] = c11[r];
        }
        __syncthreads();
        if (tid < 64) {   // online logsumexp + label pick
            float lm = -1e30f;
            #pragma unroll 8
            for (int j = 0; j < 128; ++j) lm = fmaxf(lm, tile[tid][j]);
            float m0 = rmax[tid];
            float nm = fmaxf(m0, lm);
            float s = rsum[tid] * __expf(m0 - nm);
            #pragma unroll 8
            for (int j = 0; j < 128; ++j) s += __expf(tile[tid][j] - nm);
            rmax[tid] = nm; rsum[tid] = s;
            int lab = labels[r0 + tid];
            int rel = lab - bn;
            if (rel >= 0 && rel < 128) rlab[tid] = tile[tid][rel];
        }
        __syncthreads();
    }
    if (tid < 64) {
        float nll = (rmax[tid] + __logf(rsum[tid])) - rlab[tid];
        atomicAdd(loss_accum, nll);
    }
}

// ---------------------------------------------------------------------------
// Support kernels
// ---------------------------------------------------------------------------
__global__ __launch_bounds__(256)
void rmsnorm_kernel(const float* __restrict__ x, const float* __restrict__ g,
                    float* __restrict__ out, int Hd)
{
    int row = blockIdx.x;
    const float* xr = x + (long)row * Hd;
    __shared__ float red[256];
    float s = 0.0f;
    for (int i = threadIdx.x; i < Hd; i += 256) { float v = xr[i]; s += v * v; }
    red[threadIdx.x] = s; __syncthreads();
    for (int st = 128; st > 0; st >>= 1) {
        if (threadIdx.x < st) red[threadIdx.x] += red[threadIdx.x + st];
        __syncthreads();
    }
    float scale = rsqrtf(red[0] / Hd + 1e-5f);
    for (int i = threadIdx.x; i < Hd; i += 256)
        out[(long)row * Hd + i] = xr[i] * scale * g[i];
}

__global__ __launch_bounds__(256)
void layernorm_kernel(float* __restrict__ x, const float* __restrict__ g,
                      const float* __restrict__ beta, int Hd)
{
    int row = blockIdx.x;
    float* xr = x + (long)row * Hd;
    __shared__ float red[256];
    float s = 0.0f;
    for (int i = threadIdx.x; i < Hd; i += 256) s += xr[i];
    red[threadIdx.x] = s; __syncthreads();
    for (int st = 128; st > 0; st >>= 1) {
        if (threadIdx.x < st) red[threadIdx.x] += red[threadIdx.x + st];
        __syncthreads();
    }
    float mean = red[0] / Hd; __syncthreads();
    float v = 0.0f;
    for (int i = threadIdx.x; i < Hd; i += 256) { float d = xr[i] - mean; v += d * d; }
    red[threadIdx.x] = v; __syncthreads();
    for (int st = 128; st > 0; st >>= 1) {
        if (threadIdx.x < st) red[threadIdx.x] += red[threadIdx.x + st];
        __syncthreads();
    }
    float inv = rsqrtf(red[0] / Hd + 1e-5f);
    for (int i = threadIdx.x; i < Hd; i += 256)
        xr[i] = (xr[i] - mean) * inv * g[i] + beta[i];
}

__global__ void meantime_kernel(const float* __restrict__ in, float* __restrict__ out,
                                int T, int Hd)
{
    int b = blockIdx.y;
    int h = blockIdx.x * 256 + threadIdx.x;
    if (h >= Hd) return;
    float s = 0.0f;
    for (int t = 0; t < T; ++t) s += in[((long)b * T + t) * Hd + h];
    out[(long)b * Hd + h] = s / (float)T;
}

__global__ __launch_bounds__(256)
void build_x_kernel(const int* __restrict__ ids, const float* __restrict__ embed,
                    const float* __restrict__ vemb, const float* __restrict__ aemb,
                    float* __restrict__ x)
{
    int row = blockIdx.x;       // 0..2055
    int b = row / 514, t = row % 514;
    const float* src;
    if (t == 0)      src = vemb + (long)b * 1024;
    else if (t == 1) src = aemb + (long)b * 1024;
    else             src = embed + (long)ids[b * 512 + (t - 2)] * 1024;
    for (int i = threadIdx.x; i < 1024; i += 256)
        x[(long)row * 1024 + i] = src[i];
}

__global__ __launch_bounds__(256)
void rope_kernel(float* __restrict__ q, int T)
{
    int row = blockIdx.x;
    int t = row % T;
    for (int p = threadIdx.x; p < 8 * 64; p += 256) {
        int nh = p >> 6, d = p & 63;
        float inv = __powf(10000.0f, -(float)d / 64.0f);
        float fr = (float)t * inv;
        float c = __cosf(fr), s = __sinf(fr);
        long base = (long)row * 1024 + nh * 128;
        float x1 = q[base + d], x2 = q[base + 64 + d];
        q[base + d]      = x1 * c - x2 * s;
        q[base + 64 + d] = x2 * c + x1 * s;
    }
}

__global__ __launch_bounds__(128)
void attn_kernel(const float* __restrict__ q, const float* __restrict__ k,
                 const float* __restrict__ v, const float* __restrict__ mask,
                 float* __restrict__ out, int T)
{
    int idx = blockIdx.x;
    int qpos = idx % T;
    int bh = idx / T;
    int nh = bh % 8, b = bh / 8;
    __shared__ float sc[514];
    __shared__ float qv[128];
    __shared__ float red[128];
    long qbase = ((long)b * T + qpos) * 1024 + nh * 128;
    qv[threadIdx.x] = q[qbase + threadIdx.x];
    __syncthreads();
    const float inv_sqrt = 0.08838834764831845f;  // 1/sqrt(128)
    for (int kk = threadIdx.x; kk <= qpos; kk += 128) {
        bool valid = (kk < 2) || (mask[b * 512 + (kk - 2)] > 0.5f);
        float s = 0.0f;
        const float* kr = k + ((long)b * T + kk) * 1024 + nh * 128;
        #pragma unroll 8
        for (int d = 0; d < 128; ++d) s += qv[d] * kr[d];
        sc[kk] = valid ? s * inv_sqrt : -1e9f;
    }
    __syncthreads();
    float lm = -1e30f;
    for (int kk = threadIdx.x; kk <= qpos; kk += 128) lm = fmaxf(lm, sc[kk]);
    red[threadIdx.x] = lm; __syncthreads();
    for (int st = 64; st > 0; st >>= 1) {
        if (threadIdx.x < st) red[threadIdx.x] = fmaxf(red[threadIdx.x], red[threadIdx.x + st]);
        __syncthreads();
    }
    float mx = red[0]; __syncthreads();
    float ls = 0.0f;
    for (int kk = threadIdx.x; kk <= qpos; kk += 128) {
        float e = __expf(sc[kk] - mx); sc[kk] = e; ls += e;
    }
    red[threadIdx.x] = ls; __syncthreads();
    for (int st = 64; st > 0; st >>= 1) {
        if (threadIdx.x < st) red[threadIdx.x] += red[threadIdx.x + st];
        __syncthreads();
    }
    float denom = red[0];
    int d = threadIdx.x;
    float acc = 0.0f;
    for (int kk = 0; kk <= qpos; ++kk)
        acc += sc[kk] * v[((long)b * T + kk) * 1024 + nh * 128 + d];
    out[((long)b * T + qpos) * 1024 + nh * 128 + d] = acc / denom;
}

__global__ void silu_mul_kernel(float* __restrict__ g, const float* __restrict__ u, int n)
{
    int i = blockIdx.x * 256 + threadIdx.x;
    if (i < n) {
        float gv = g[i];
        float s = gv / (1.0f + __expf(-gv));
        g[i] = s * u[i];
    }
}

__device__ float block_dot(const float* a, const float* b, int n, float* red)
{
    float s = 0.0f;
    for (int i = threadIdx.x; i < n; i += 256) s += a[i] * b[i];
    red[threadIdx.x] = s; __syncthreads();
    for (int st = 128; st > 0; st >>= 1) {
        if (threadIdx.x < st) red[threadIdx.x] += red[threadIdx.x + st];
        __syncthreads();
    }
    float r = red[0]; __syncthreads();
    return r;
}

__global__ __launch_bounds__(256)
void contrast_kernel(const float* __restrict__ vemb, const float* __restrict__ aemb,
                     const float* __restrict__ embed, const int* __restrict__ ids,
                     const float* __restrict__ temp_p, float* __restrict__ out_scalar)
{
    __shared__ float red[256];
    __shared__ float invv[4], inva[4], invt[4];
    __shared__ float vt[4][4], at[4][4];
    for (int b = 0; b < 4; ++b) {
        float d = block_dot(vemb + b * 1024, vemb + b * 1024, 1024, red);
        if (threadIdx.x == 0) invv[b] = 1.0f / fmaxf(sqrtf(d), 1e-6f);
        d = block_dot(aemb + b * 1024, aemb + b * 1024, 1024, red);
        if (threadIdx.x == 0) inva[b] = 1.0f / fmaxf(sqrtf(d), 1e-6f);
        const float* t = embed + (long)ids[b * 512] * 1024;
        d = block_dot(t, t, 1024, red);
        if (threadIdx.x == 0) invt[b] = 1.0f / fmaxf(sqrtf(d), 1e-6f);
    }
    __syncthreads();
    float temp = fminf(fmaxf(temp_p[0], 0.01f), 0.5f);
    for (int i = 0; i < 4; ++i)
        for (int j = 0; j < 4; ++j) {
            const float* t = embed + (long)ids[j * 512] * 1024;
            float dv = block_dot(vemb + i * 1024, t, 1024, red);
            float da = block_dot(aemb + i * 1024, t, 1024, red);
            if (threadIdx.x == 0) {
                vt[i][j] = fminf(dv * invv[i] * invt[j] / temp, 30.0f);
                at[i][j] = fminf(da * inva[i] * invt[j] / temp, 30.0f);
            }
        }
    __syncthreads();
    if (threadIdx.x == 0) {
        float closs = 0.0f;
        for (int pass = 0; pass < 2; ++pass) {
            float (*S)[4] = pass ? at : vt;
            float acc = 0.0f;
            for (int i = 0; i < 4; ++i) {
                float mx = S[i][0];
                for (int j = 1; j < 4; ++j) mx = fmaxf(mx, S[i][j]);
                float se = 0.0f;
                for (int j = 0; j < 4; ++j) se += __expf(S[i][j] - mx);
                acc += (mx + __logf(se)) - S[i][i];
            }
            closs += 0.5f * (acc / 4.0f);
        }
        out_scalar[0] = closs;
    }
}

__global__ void zero_kernel(float* p, int n)
{
    int i = blockIdx.x * 64 + threadIdx.x;
    if (i < n) p[i] = 0.0f;
}

__global__ void finalize_kernel(const float* lm_sum, const float* closs, float* out)
{
    if (threadIdx.x == 0) out[0] = lm_sum[0] / 2048.0f + 0.25f * closs[0];
}

// ---------------------------------------------------------------------------
extern "C" void kernel_launch(void* const* d_in, const int* in_sizes, int n_in,
                              void* d_out, int out_size, void* d_ws, size_t ws_size,
                              hipStream_t stream)
{
    (void)in_sizes; (void)n_in; (void)out_size; (void)ws_size;
    const int*   input_ids = (const int*)  d_in[0];
    const float* amask     = (const float*)d_in[1];
    const int*   labels    = (const int*)  d_in[2];
    const float* visual    = (const float*)d_in[3];
    const float* acoustic  = (const float*)d_in[4];
    const float* embed     = (const float*)d_in[5];
    const float* vp_w1 = (const float*)d_in[6];
    const float* vp_b1 = (const float*)d_in[7];
    const float* vp_w2 = (const float*)d_in[8];
    const float* vp_b2 = (const float*)d_in[9];
    const float* vp_g  = (const float*)d_in[10];
    const float* vp_be = (const float*)d_in[11];
    const float* ap_w1 = (const float*)d_in[12];
    const float* ap_b1 = (const float*)d_in[13];
    const float* ap_w2 = (const float*)d_in[14];
    const float* ap_b2 = (const float*)d_in[15];
    const float* ap_g  = (const float*)d_in[16];
    const float* ap_be = (const float*)d_in[17];
    const float* temp  = (const float*)d_in[18];
    const float* wq = (const float*)d_in[19];
    const float* aq = (const float*)d_in[20];
    const float* bq = (const float*)d_in[21];
    const float* wk = (const float*)d_in[22];
    const float* ak = (const float*)d_in[23];
    const float* bk = (const float*)d_in[24];
    const float* wv = (const float*)d_in[25];
    const float* av = (const float*)d_in[26];
    const float* bv = (const float*)d_in[27];
    const float* wo = (const float*)d_in[28];
    const float* ao = (const float*)d_in[29];
    const float* bo = (const float*)d_in[30];
    const float* wg = (const float*)d_in[31];
    const float* wu = (const float*)d_in[32];
    const float* wd = (const float*)d_in[33];
    const float* ln1 = (const float*)d_in[34];
    const float* ln2 = (const float*)d_in[35];
    const float* lnf = (const float*)d_in[36];
    const float* lm_head = (const float*)d_in[37];

    const int T = 514, ROWS = 4 * T;   // 2056
    float* ws = (float*)d_ws;
    size_t off = 0;
    auto alloc = [&](size_t n) { float* p = ws + off; off += n; return p; };
    float* x     = alloc((size_t)ROWS * 1024);
    float* h     = alloc((size_t)ROWS * 1024);
    float* qb    = alloc((size_t)ROWS * 1024);
    float* kb    = alloc((size_t)ROWS * 1024);
    float* vb    = alloc((size_t)ROWS * 1024);
    float* ab    = alloc((size_t)ROWS * 1024);     // attn out
    float* gbuf  = alloc((size_t)ROWS * 4096);
    float* ubuf  = alloc((size_t)ROWS * 4096);
    float* ltmp  = alloc((size_t)ROWS * 32);
    float* vh    = alloc((size_t)256 * 2048);
    float* vproj = alloc((size_t)256 * 1024);
    float* ah    = alloc((size_t)512 * 1024);
    float* aproj = alloc((size_t)512 * 1024);
    float* vemb  = alloc(4 * 1024);
    float* aemb  = alloc(4 * 1024);
    float* scal  = alloc(8);   // [0]=lm nll sum, [1]=contrast loss

    auto gemm = [&](const float* A, int lda, const float* B, int ldb, const float* bias,
                    float* C, int ldc, int M, int N, int K, float alpha, int act, int acc) {
        dim3 grid((N + 127) / 128, (M + 63) / 64);
        gemm_wmma_kernel<<<grid, 256, 0, stream>>>(A, lda, B, ldb, bias, C, ldc,
                                                   M, N, K, alpha, act, acc);
    };

    zero_kernel<<<1, 64, 0, stream>>>(scal, 8);

    // ---- projectors ----
    gemm(visual, 1024, vp_w1, 2048, vp_b1, vh, 2048, 256, 2048, 1024, 1.0f, ACT_RELU, 0);
    gemm(vh, 2048, vp_w2, 1024, vp_b2, vproj, 1024, 256, 1024, 2048, 1.0f, ACT_NONE, 0);
    layernorm_kernel<<<256, 256, 0, stream>>>(vproj, vp_g, vp_be, 1024);
    meantime_kernel<<<dim3(4, 4), 256, 0, stream>>>(vproj, vemb, 64, 1024);

    gemm(acoustic, 512, ap_w1, 1024, ap_b1, ah, 1024, 512, 1024, 512, 1.0f, ACT_RELU, 0);
    gemm(ah, 1024, ap_w2, 1024, ap_b2, aproj, 1024, 512, 1024, 1024, 1.0f, ACT_NONE, 0);
    layernorm_kernel<<<512, 256, 0, stream>>>(aproj, ap_g, ap_be, 1024);
    meantime_kernel<<<dim3(4, 4), 256, 0, stream>>>(aproj, aemb, 128, 1024);

    // ---- build sequence ----
    build_x_kernel<<<ROWS, 256, 0, stream>>>(input_ids, embed, vemb, aemb, x);

    // ---- transformer layers ----
    const float LS = 2.0f;   // LoRA scale 64/32
    for (int l = 0; l < 2; ++l) {
        const float* wq_l = wq + (size_t)l * 1024 * 1024;
        const float* aq_l = aq + (size_t)l * 1024 * 32;
        const float* bq_l = bq + (size_t)l * 32 * 1024;
        const float* wk_l = wk + (size_t)l * 1024 * 1024;
        const float* ak_l = ak + (size_t)l * 1024 * 32;
        const float* bk_l = bk + (size_t)l * 32 * 1024;
        const float* wv_l = wv + (size_t)l * 1024 * 1024;
        const float* av_l = av + (size_t)l * 1024 * 32;
        const float* bv_l = bv + (size_t)l * 32 * 1024;
        const float* wo_l = wo + (size_t)l * 1024 * 1024;
        const float* ao_l = ao + (size_t)l * 1024 * 32;
        const float* bo_l = bo + (size_t)l * 32 * 1024;
        const float* wg_l = wg + (size_t)l * 1024 * 4096;
        const float* wu_l = wu + (size_t)l * 1024 * 4096;
        const float* wd_l = wd + (size_t)l * 4096 * 1024;

        rmsnorm_kernel<<<ROWS, 256, 0, stream>>>(x, ln1 + l * 1024, h, 1024);

        gemm(h, 1024, wq_l, 1024, nullptr, qb, 1024, ROWS, 1024, 1024, 1.0f, ACT_NONE, 0);
        gemm(h, 1024, aq_l, 32, nullptr, ltmp, 32, ROWS, 32, 1024, 1.0f, ACT_NONE, 0);
        gemm(ltmp, 32, bq_l, 1024, nullptr, qb, 1024, ROWS, 1024, 32, LS, ACT_NONE, 1);

        gemm(h, 1024, wk_l, 1024, nullptr, kb, 1024, ROWS, 1024, 1024, 1.0f, ACT_NONE, 0);
        gemm(h, 1024, ak_l, 32, nullptr, ltmp, 32, ROWS, 32, 1024, 1.0f, ACT_NONE, 0);
        gemm(ltmp, 32, bk_l, 1024, nullptr, kb, 1024, ROWS, 1024, 32, LS, ACT_NONE, 1);

        gemm(h, 1024, wv_l, 1024, nullptr, vb, 1024, ROWS, 1024, 1024, 1.0f, ACT_NONE, 0);
        gemm(h, 1024, av_l, 32, nullptr, ltmp, 32, ROWS, 32, 1024, 1.0f, ACT_NONE, 0);
        gemm(ltmp, 32, bv_l, 1024, nullptr, vb, 1024, ROWS, 1024, 32, LS, ACT_NONE, 1);

        rope_kernel<<<ROWS, 256, 0, stream>>>(qb, T);
        rope_kernel<<<ROWS, 256, 0, stream>>>(kb, T);

        attn_kernel<<<4 * 8 * T, 128, 0, stream>>>(qb, kb, vb, amask, ab, T);

        gemm(ab, 1024, wo_l, 1024, nullptr, x, 1024, ROWS, 1024, 1024, 1.0f, ACT_NONE, 1);
        gemm(ab, 1024, ao_l, 32, nullptr, ltmp, 32, ROWS, 32, 1024, 1.0f, ACT_NONE, 0);
        gemm(ltmp, 32, bo_l, 1024, nullptr, x, 1024, ROWS, 1024, 32, LS, ACT_NONE, 1);

        rmsnorm_kernel<<<ROWS, 256, 0, stream>>>(x, ln2 + l * 1024, h, 1024);
        gemm(h, 1024, wg_l, 4096, nullptr, gbuf, 4096, ROWS, 4096, 1024, 1.0f, ACT_NONE, 0);
        gemm(h, 1024, wu_l, 4096, nullptr, ubuf, 4096, ROWS, 4096, 1024, 1.0f, ACT_NONE, 0);
        silu_mul_kernel<<<(ROWS * 4096 + 255) / 256, 256, 0, stream>>>(gbuf, ubuf, ROWS * 4096);
        gemm(gbuf, 4096, wd_l, 1024, nullptr, x, 1024, ROWS, 1024, 4096, 1.0f, ACT_NONE, 1);
    }

    // ---- final norm + fused LM head / CE ----
    rmsnorm_kernel<<<ROWS, 256, 0, stream>>>(x, lnf, h, 1024);
    lmhead_loss_kernel<<<32, 256, 0, stream>>>(h, lm_head, labels, scal);

    // ---- contrastive loss + finalize ----
    contrast_kernel<<<1, 256, 0, stream>>>(vemb, aemb, embed, input_ids, temp, scal + 1);
    finalize_kernel<<<1, 32, 0, stream>>>(scal, scal + 1, (float*)d_out);
}
